// GraphAttentionLayer_10574209483441
// MI455X (gfx1250) — matrix operand order repaired and checked by reading, not compiled, not run on hardware
//
#include <hip/hip_runtime.h>
#include <hip/hip_bf16.h>

typedef __attribute__((ext_vector_type(16))) __bf16 v16bf;
typedef __attribute__((ext_vector_type(8)))  float  v8f;

#define ALPHA 0.2f

union BfTile {
    v16bf v;
    unsigned u[8];
    uint4 q[2];
};

__device__ __forceinline__ unsigned short f2bf(float f) {
    union { float f; unsigned u; } v; v.f = f;
    unsigned u = v.u;
    u += 0x7fffu + ((u >> 16) & 1u);      // round-to-nearest-even
    return (unsigned short)(u >> 16);
}
__device__ __forceinline__ float bf2f(unsigned short h) {
    union { unsigned u; float f; } v; v.u = ((unsigned)h) << 16;
    return v.f;
}
__device__ __forceinline__ unsigned pack2(float lo, float hi) {
    return (unsigned)f2bf(lo) | ((unsigned)f2bf(hi) << 16);
}

// ---------------------------------------------------------------------------
// Kernel 0: convert h -> bf16 (row major [16384][256]); W -> WT bf16 [e][k]
// ---------------------------------------------------------------------------
__global__ void gat_prep(const float* __restrict__ h, const float* __restrict__ W,
                         unsigned short* __restrict__ hbf, unsigned short* __restrict__ WT,
                         int n_h) {
    int t = blockIdx.x * blockDim.x + threadIdx.x;
    if (t < n_h) hbf[t] = f2bf(h[t]);
    if (t < 256 * 128) {                 // W is (256,128) row-major
        int k = t >> 7, e = t & 127;
        WT[e * 256 + k] = f2bf(W[t]);
    }
}

// ---------------------------------------------------------------------------
// Kernel 1: Wh = h @ W via bf16 WMMA (f32 accumulate).
// Writes WhT bf16 laid out [b][e][j] so the attention kernel's B-matrix loads
// are contiguous. 8 waves/block, 1 wave = one 16-row tile, full 128 columns.
// ---------------------------------------------------------------------------
__global__ __launch_bounds__(256) void gat_gemm_wh(
        const unsigned short* __restrict__ hbf,
        const unsigned short* __restrict__ WT,
        unsigned short* __restrict__ WhT) {
    const int lane = threadIdx.x & 31;
    const int wv   = threadIdx.x >> 5;
    const int l15  = lane & 15;
    const int hi   = (lane >> 4) & 1;
    const int i0   = (blockIdx.x * 8 + wv) * 16;   // global row (b*2048+i), 0..16383
    const int b    = i0 >> 11;
    const int jj0  = i0 & 2047;

    v8f acc[8] = {};

    const unsigned short* hrow = hbf + (size_t)(i0 + l15) * 256;

    for (int ks = 0; ks < 8; ++ks) {
        const int k0 = ks * 32;
        // A: 16x32 bf16. lane<16: row=lane, K {0..7} -> A0..3, {16..23} -> A4..7
        //                lane>=16: K {8..15} / {24..31}
        BfTile A;
        const unsigned short* pa = hrow + k0 + hi * 8;
        A.q[0] = *(const uint4*)(pa);
        A.q[1] = *(const uint4*)(pa + 16);
        #pragma unroll
        for (int t = 0; t < 8; ++t) {
            const int e0 = t * 16;
            // B: 32x16 bf16. lane<16: col=lane, K {0..7},{16..23}; lanes>=16: K {8..15},{24..31}
            BfTile Bm;
            const unsigned short* pb = WT + (size_t)(e0 + l15) * 256 + k0 + hi * 8;
            Bm.q[0] = *(const uint4*)(pb);
            Bm.q[1] = *(const uint4*)(pb + 16);
            acc[t] = __builtin_amdgcn_wmma_f32_16x16x32_bf16(
                false, A.v, false, Bm.v, (short)0, acc[t], false, false);
        }
    }

    // C layout: VGPR r, lane<16 -> row M=r, col N=lane; lane>=16 -> M=r+8.
    // Store to WhT[b][e][j]: per (tile,lane) the 8 values are j-contiguous.
    #pragma unroll
    for (int t = 0; t < 8; ++t) {
        const int e = t * 16 + l15;
        unsigned short* base = WhT + ((size_t)(b * 128 + e) << 11) + jj0 + hi * 8;
        uint4 st;
        st.x = pack2(acc[t][0], acc[t][1]);
        st.y = pack2(acc[t][2], acc[t][3]);
        st.z = pack2(acc[t][4], acc[t][5]);
        st.w = pack2(acc[t][6], acc[t][7]);
        *(uint4*)base = st;
    }
}

// ---------------------------------------------------------------------------
// Kernel 2: s_src/s_dst = Wh @ a_src / a_dst  (one thread per (b,j))
// ---------------------------------------------------------------------------
__global__ __launch_bounds__(256) void gat_scores(
        const unsigned short* __restrict__ WhT, const float* __restrict__ a,
        float* __restrict__ s_src, float* __restrict__ s_dst) {
    const int t = blockIdx.x * blockDim.x + threadIdx.x;   // 0..16383
    const int b = t >> 11, j = t & 2047;
    const unsigned short* base = WhT + ((size_t)(b * 128) << 11) + j;
    float as = 0.f, ad = 0.f;
    for (int e = 0; e < 128; ++e) {
        const float w = bf2f(base[(size_t)e << 11]);
        as += w * a[e];
        ad += w * a[128 + e];
    }
    s_src[t] = as;
    s_dst[t] = ad;
}

// ---------------------------------------------------------------------------
// Kernel 3: fused mask + softmax + (attention @ Wh) + ELU.
// 1 block = one 16-query tile; 4 waves, wave w owns output columns [32w,32w+32).
// P (16x32 attention probs) built directly in the bf16 A-register layout.
// Software-pipelined: adj/s_dst for block j+32 are issued before the current
// block's exp chain + WMMAs, so the HBM-latency adj stream runs one iteration
// ahead (WMMA then only waits on the L2-resident Wh loads).
// ---------------------------------------------------------------------------
__global__ __launch_bounds__(128) void gat_attn(
        const int* __restrict__ adj, const unsigned short* __restrict__ WhT,
        const float* __restrict__ s_src, const float* __restrict__ s_dst,
        float* __restrict__ out) {
    const int lane = threadIdx.x & 31;
    const int wv   = threadIdx.x >> 5;        // 0..3
    const int l15  = lane & 15;
    const int hi   = (lane >> 4) & 1;
    const int hi8  = hi * 8;
    const int i0g  = blockIdx.x * 16;         // global row base (b*2048 + i0)
    const int b    = i0g >> 11;

    const int*   adjRow = adj + ((size_t)(i0g + l15) << 11);
    const float* sdB    = s_dst + ((size_t)b << 11);
    const float  ssrc   = s_src[i0g + l15];

    const int e0a = wv * 32;
    const unsigned short* wbase0 =
        WhT + ((size_t)(b * 128 + e0a + l15) << 11) + hi * 8;
    const unsigned short* wbase1 = wbase0 + ((size_t)16 << 11);

    v8f acc0 = {}, acc1 = {};
    float rowsum = 0.f;

    // ---- pipeline prologue: load block j0 = 0 ----
    int4   qc0 = *(const int4*)(adjRow + hi8);
    int4   qc1 = *(const int4*)(adjRow + hi8 + 4);
    int4   qc2 = *(const int4*)(adjRow + hi8 + 16);
    int4   qc3 = *(const int4*)(adjRow + hi8 + 20);
    float4 sc0 = *(const float4*)(sdB + hi8);
    float4 sc1 = *(const float4*)(sdB + hi8 + 4);
    float4 sc2 = *(const float4*)(sdB + hi8 + 16);
    float4 sc3 = *(const float4*)(sdB + hi8 + 20);

    for (int j0 = 0; j0 < 2048; j0 += 32) {
        // ---- issue NEXT block's adj/s_dst loads first (wraps on last iter,
        //      values unused; keeps speculative loads in-bounds) ----
        const int jn  = (j0 + 32 < 2048) ? (j0 + 32) : 0;
        const int kbn = jn + hi8;
        const int4   qn0 = *(const int4*)(adjRow + kbn);
        const int4   qn1 = *(const int4*)(adjRow + kbn + 4);
        const int4   qn2 = *(const int4*)(adjRow + kbn + 16);
        const int4   qn3 = *(const int4*)(adjRow + kbn + 20);
        const float4 sn0 = *(const float4*)(sdB + kbn);
        const float4 sn1 = *(const float4*)(sdB + kbn + 4);
        const float4 sn2 = *(const float4*)(sdB + kbn + 16);
        const float4 sn3 = *(const float4*)(sdB + kbn + 20);
        __builtin_prefetch(adjRow + kbn + 32, 0, 1);   // two blocks ahead

        // ---- B matrices for the CURRENT block (L2-resident Wh) ----
        BfTile B0, B1;
        const unsigned short* pb0 = wbase0 + j0;
        B0.q[0] = *(const uint4*)(pb0);
        B0.q[1] = *(const uint4*)(pb0 + 16);
        const unsigned short* pb1 = wbase1 + j0;
        B1.q[0] = *(const uint4*)(pb1);
        B1.q[1] = *(const uint4*)(pb1 + 16);

        // p = adj>0 ? exp(leakyrelu(ssrc + sdst)) : 0   (logits are O(10): no
        // overflow possible in fp32 -> no running-max rescale needed)
        #define PE(av, sv) ({ float _x = ssrc + (sv);                      \
                              _x = fmaxf(_x, _x * ALPHA);                  \
                              float _e = __expf(_x);                       \
                              ((av) > 0) ? _e : 0.f; })
        float p0  = PE(qc0.x, sc0.x), p1  = PE(qc0.y, sc0.y);
        float p2  = PE(qc0.z, sc0.z), p3  = PE(qc0.w, sc0.w);
        float p4  = PE(qc1.x, sc1.x), p5  = PE(qc1.y, sc1.y);
        float p6  = PE(qc1.z, sc1.z), p7  = PE(qc1.w, sc1.w);
        float p8  = PE(qc2.x, sc2.x), p9  = PE(qc2.y, sc2.y);
        float p10 = PE(qc2.z, sc2.z), p11 = PE(qc2.w, sc2.w);
        float p12 = PE(qc3.x, sc3.x), p13 = PE(qc3.y, sc3.y);
        float p14 = PE(qc3.z, sc3.z), p15 = PE(qc3.w, sc3.w);
        #undef PE

        rowsum += ((p0 + p1) + (p2 + p3)) + ((p4 + p5) + (p6 + p7)) +
                  ((p8 + p9) + (p10 + p11)) + ((p12 + p13) + (p14 + p15));

        BfTile P;
        P.u[0] = pack2(p0,  p1);  P.u[1] = pack2(p2,  p3);
        P.u[2] = pack2(p4,  p5);  P.u[3] = pack2(p6,  p7);
        P.u[4] = pack2(p8,  p9);  P.u[5] = pack2(p10, p11);
        P.u[6] = pack2(p12, p13); P.u[7] = pack2(p14, p15);

        acc0 = __builtin_amdgcn_wmma_f32_16x16x32_bf16(
            false, P.v, false, B0.v, (short)0, acc0, false, false);
        acc1 = __builtin_amdgcn_wmma_f32_16x16x32_bf16(
            false, P.v, false, B1.v, (short)0, acc1, false, false);

        // ---- rotate pipeline registers ----
        qc0 = qn0; qc1 = qn1; qc2 = qn2; qc3 = qn3;
        sc0 = sn0; sc1 = sn1; sc2 = sn2; sc3 = sn3;
    }

    // lane l and l^16 each hold half of row (l&15)'s sum
    const float rs = rowsum + __shfl_xor(rowsum, 16, 32);

    #pragma unroll
    for (int r = 0; r < 8; ++r) {
        const int   srcLane = r + hi * 8;              // row index M = r + hi*8
        const float denom   = __shfl(rs, srcLane, 32);
        const float inv     = denom > 0.f ? 1.0f / denom : 0.f;
        float o0 = acc0[r] * inv;
        float o1 = acc1[r] * inv;
        o0 = o0 > 0.f ? o0 : __expf(o0) - 1.f;         // ELU(alpha=1)
        o1 = o1 > 0.f ? o1 : __expf(o1) - 1.f;
        const size_t rowOff = (size_t)(i0g + r + hi * 8) * 128;
        out[rowOff + e0a + l15]      = o0;
        out[rowOff + e0a + 16 + l15] = o1;
    }
}

// ---------------------------------------------------------------------------
extern "C" void kernel_launch(void* const* d_in, const int* in_sizes, int n_in,
                              void* d_out, int out_size, void* d_ws, size_t ws_size,
                              hipStream_t stream) {
    const float* h   = (const float*)d_in[0];   // (8,2048,256)
    const int*   adj = (const int*)  d_in[1];   // (8,2048,2048)
    const float* W   = (const float*)d_in[2];   // (256,128)
    const float* a   = (const float*)d_in[3];   // (256,1)
    float* out = (float*)d_out;                 // (8,2048,128)

    char* ws = (char*)d_ws;
    const size_t N_H  = 8ull * 2048 * 256;      // 4,194,304
    unsigned short* hbf  = (unsigned short*)(ws);                       // 8 MB
    unsigned short* WT   = (unsigned short*)(ws + 8u * 1024 * 1024);    // 64 KB
    unsigned short* WhT  = (unsigned short*)(ws + 8u * 1024 * 1024 + 65536); // 4 MB
    float* s_src = (float*)(ws + 12u * 1024 * 1024 + 65536);            // 64 KB
    float* s_dst = (float*)(ws + 12u * 1024 * 1024 + 131072);           // 64 KB

    gat_prep<<<(int)((N_H + 255) / 256), 256, 0, stream>>>(h, W, hbf, WT, (int)N_H);
    gat_gemm_wh<<<128, 256, 0, stream>>>(hbf, WT, WhT);
    gat_scores<<<64, 256, 0, stream>>>(WhT, a, s_src, s_dst);
    gat_attn<<<1024, 128, 0, stream>>>(adj, WhT, s_src, s_dst, out);
}
